// MultiHeadedAttention_257698038597
// MI455X (gfx1250) — matrix operand order repaired and checked by reading, not compiled, not run on hardware
//
#include <hip/hip_runtime.h>

// ---------------- types ----------------
typedef __bf16 bf16;
typedef __attribute__((ext_vector_type(16))) __bf16 v16bf;
typedef __attribute__((ext_vector_type(8)))  float  v8f;
typedef __attribute__((ext_vector_type(4)))  int    v4i;

__device__ __forceinline__ bf16 f2bf(float f) {
    union { float f; unsigned u; } v; v.f = f;
    unsigned r = (v.u + 0x7FFFu + ((v.u >> 16) & 1u)) >> 16;
    union { unsigned short s; bf16 b; } o; o.s = (unsigned short)r;
    return o.b;
}

// ---------------- async global->LDS 16B copy (CDNA5 path w/ fallback) ----------------
__device__ __forceinline__ void copy16_g2s(bf16* __restrict__ dst, const bf16* __restrict__ src) {
#if defined(__gfx1250__) && __has_builtin(__builtin_amdgcn_global_load_async_to_lds_b128)
    __builtin_amdgcn_global_load_async_to_lds_b128(
        (__attribute__((address_space(1))) v4i*)src,
        (__attribute__((address_space(3))) v4i*)dst, 0, 0);
#else
    *(uint4*)dst = *(const uint4*)src;
#endif
}

__device__ __forceinline__ void async_fence() {
#if defined(__gfx1250__) && __has_builtin(__builtin_amdgcn_s_wait_asynccnt)
    __builtin_amdgcn_s_wait_asynccnt(0);
#endif
}

// ---------------- WMMA fragment gathers (contiguous k = 16*half + j, same order in A and B) ----------------
// A fragment 16x32 (MxK) bf16 from LDS row-major [m][k], row stride ld elements.
__device__ __forceinline__ v16bf load_frag_a(const bf16* __restrict__ s, int ld) {
    int lane = threadIdx.x & 31;
    int r    = lane & 15;
    int hf   = lane >> 4;
    const bf16* row = s + r * ld;
    v16bf a;
#pragma unroll
    for (int j = 0; j < 16; ++j) {
        int k = j + 16 * hf;
        a[j] = row[k];
    }
    return a;
}

// B fragment 32x16 (KxN) bf16 from LDS stored TRANSPOSED: sT[n][k], stride ld.
__device__ __forceinline__ v16bf load_frag_b(const bf16* __restrict__ sT, int ld) {
    int lane = threadIdx.x & 31;
    int n    = lane & 15;
    int hf   = lane >> 4;
    const bf16* row = sT + n * ld + 16 * hf;
    v16bf b;
#pragma unroll
    for (int j = 0; j < 16; ++j) b[j] = row[j];
    return b;
}

// ---------------- f32 -> bf16 conversion ----------------
__global__ void cvt_f32_bf16(const float* __restrict__ s, bf16* __restrict__ d, int n) {
    int i = (blockIdx.x * blockDim.x + threadIdx.x) * 4;
    if (i + 3 < n) {
        float4 f = *(const float4*)(s + i);
        d[i + 0] = f2bf(f.x);
        d[i + 1] = f2bf(f.y);
        d[i + 2] = f2bf(f.z);
        d[i + 3] = f2bf(f.w);
    }
}

// ---------------- f32 W[512][512] -> bf16 W^T[512][512] (one-time) ----------------
__global__ __launch_bounds__(256) void cvt_transpose_f32_bf16(
    const float* __restrict__ W, bf16* __restrict__ WT) {
    __shared__ float tile[32][33];
    const int bx = blockIdx.x * 32;   // k block (rows of W)
    const int by = blockIdx.y * 32;   // n block (cols of W)
    const int tx = threadIdx.x & 31, ty = threadIdx.x >> 5;   // 32x8
#pragma unroll
    for (int i = 0; i < 32; i += 8)
        tile[ty + i][tx] = W[(size_t)(bx + ty + i) * 512 + by + tx];
    __syncthreads();
#pragma unroll
    for (int i = 0; i < 32; i += 8)
        WT[(size_t)(by + ty + i) * 512 + bx + tx] = f2bf(tile[tx][ty + i]);
}

// ---------------- GEMM: out = A(8192x512) @ W(512x512) + bias ----------------
// WT is the pre-transposed bf16 weight: WT[n][k].
// mode 0: out bf16 [b][h][c][t][dk]   (Q/K projection)
// mode 2: out bf16 [b][h][c][dk][t]   (V projection, pre-transposed for attention)
// mode 1: out f32 row-major [m][n]    (output projection)
__global__ __launch_bounds__(256) void gemm_bf16(
    const bf16* __restrict__ A, const bf16* __restrict__ WT,
    const float* __restrict__ bias, void* __restrict__ out, int mode)
{
    constexpr int K = 512, TM = 128, TN = 128, KS = 32;
    __shared__ __align__(16) bf16 As[TM * KS];   // [m][k], ld = 32
    __shared__ __align__(16) bf16 BsT[TN * KS];  // [n][k], ld = 32

    const int row0 = blockIdx.x * TM;
    const int col0 = blockIdx.y * TN;
    const int tid  = threadIdx.x;
    const int wave = tid >> 5;
    const int wm   = (wave & 3) * 32;    // 4 M strips of 32
    const int wn   = (wave >> 2) * 64;   // 2 N strips of 64

    v8f acc[2][4] = {};

    for (int k0 = 0; k0 < K; k0 += KS) {
        // stage A tile: 128x32 bf16, 16 elems (2 x 16B) per thread
        {
            int m = tid >> 1, seg = (tid & 1) * 16;
            const bf16* src = A + (size_t)(row0 + m) * K + k0 + seg;
            bf16* dst = As + m * KS + seg;
            copy16_g2s(dst, src);
            copy16_g2s(dst + 8, src + 8);
        }
        // stage W^T tile: 128x32 bf16, 16 elems (2 x 16B) per thread
        {
            int n = tid >> 1, seg = (tid & 1) * 16;
            const bf16* src = WT + (size_t)(col0 + n) * K + k0 + seg;
            bf16* dst = BsT + n * KS + seg;
            copy16_g2s(dst, src);
            copy16_g2s(dst + 8, src + 8);
        }
        if (k0 + KS < K)
            __builtin_prefetch(A + (size_t)(row0 + (tid >> 1)) * K + k0 + KS, 0, 1);
        async_fence();
        __syncthreads();

        v16bf af[2], bfr[4];
#pragma unroll
        for (int i = 0; i < 2; ++i) af[i]  = load_frag_a(As  + (wm + i * 16) * KS, KS);
#pragma unroll
        for (int j = 0; j < 4; ++j) bfr[j] = load_frag_b(BsT + (wn + j * 16) * KS, KS);
#pragma unroll
        for (int i = 0; i < 2; ++i)
#pragma unroll
            for (int j = 0; j < 4; ++j)
                acc[i][j] = __builtin_amdgcn_wmma_f32_16x16x32_bf16(
                    false, af[i], false, bfr[j], (short)0, acc[i][j], false, false);
        __syncthreads();
    }

    // epilogue: C/D layout: element r -> m = r + 8*(lane>=16), n = lane&15
    const int lane = tid & 31, hf = lane >> 4, nl = lane & 15;
#pragma unroll
    for (int i = 0; i < 2; ++i)
#pragma unroll
        for (int j = 0; j < 4; ++j)
#pragma unroll
            for (int r = 0; r < 8; ++r) {
                int m = row0 + wm + i * 16 + r + 8 * hf;
                int n = col0 + wn + j * 16 + nl;
                float vv = acc[i][j][r] + bias[n];
                int b = m >> 12, t = (m >> 3) & 511, c = m & 7;
                int h = n >> 6,  d = n & 63;
                if (mode == 0) {
                    ((bf16*)out)[((((size_t)b * 8 + h) * 8 + c) * 512 + t) * 64 + d] = f2bf(vv);
                } else if (mode == 2) {
                    ((bf16*)out)[((((size_t)b * 8 + h) * 8 + c) * 64 + d) * 512 + t] = f2bf(vv);
                } else {
                    ((float*)out)[(size_t)m * 512 + n] = vv;
                }
            }
}

// ---------------- flash-style attention over one (b,h,c, t-tile) ----------------
// Q,K layout [combo][t][dk]; V layout [combo][dk][t] (pre-transposed by mode-2 GEMM).
__global__ __launch_bounds__(256) void attn_kernel(
    const bf16* __restrict__ Q, const bf16* __restrict__ Kp, const bf16* __restrict__ V,
    const int* __restrict__ mask, bf16* __restrict__ X)
{
    constexpr int T = 512, DK = 64;
    __shared__ __align__(16) bf16 Qs [64 * 64];
    __shared__ __align__(16) bf16 Ks [64 * 64];   // [s][d] == transposed-B for QK^T
    __shared__ __align__(16) bf16 VsT[64 * 64];   // [d][s] == transposed-B for P@V
    __shared__ float Ss[64 * 64];
    __shared__ __align__(16) bf16 Ps [64 * 64];
    __shared__ float rowm[64], rowl[64], rowalpha[64];
    __shared__ float red[64 * 4];

    const int blk   = blockIdx.x;
    const int combo = blk >> 3;            // (b*8+h)*8+c
    const int t0    = (blk & 7) * 64;
    const int b = combo >> 6, h = (combo >> 3) & 7, c = combo & 7;

    const bf16* qb = Q  + (size_t)combo * T * DK;
    const bf16* kb = Kp + (size_t)combo * T * DK;
    const bf16* vb = V  + (size_t)combo * DK * T;   // [d][t]

    const int tid = threadIdx.x;
    const int wave = tid >> 5, lane = tid & 31, hf = lane >> 4, nl = lane & 15;
    const int fm  = wave >> 1;             // 0..3  (16-row strip)
    const int fn0 = (wave & 1) * 2;        // 0 or 2 (two 16-col strips)
    const float scale = 0.125f;            // 1/sqrt(64)

    // load Q tile once (async)
    {
        int r = tid >> 2, seg = (tid & 3) * 16;
        copy16_g2s(Qs + r * 64 + seg,     qb + (size_t)(t0 + r) * DK + seg);
        copy16_g2s(Qs + r * 64 + seg + 8, qb + (size_t)(t0 + r) * DK + seg + 8);
    }
    if (tid < 64) { rowm[tid] = -3.0e38f; rowl[tid] = 0.0f; }

    v8f o[2] = {};

    for (int s0 = 0; s0 < T; s0 += 64) {
        __syncthreads();   // protect Ks/VsT/Ps from previous iteration readers
        // stage K tile [s][d] and V tile [d][s] (both contiguous 16B chunks)
        {
            int r = tid >> 2, seg = (tid & 3) * 16;
            copy16_g2s(Ks + r * 64 + seg,     kb + (size_t)(s0 + r) * DK + seg);
            copy16_g2s(Ks + r * 64 + seg + 8, kb + (size_t)(s0 + r) * DK + seg + 8);
            copy16_g2s(VsT + r * 64 + seg,     vb + (size_t)r * T + s0 + seg);
            copy16_g2s(VsT + r * 64 + seg + 8, vb + (size_t)r * T + s0 + seg + 8);
        }
        async_fence();
        __syncthreads();

        // S = Q @ K^T  (64x64 tile; each wave: 1x2 fragments)
        {
            v8f sacc[2] = {};
#pragma unroll
            for (int d0 = 0; d0 < 64; d0 += 32) {
                v16bf aq = load_frag_a(Qs + fm * 16 * 64 + d0, 64);
#pragma unroll
                for (int j = 0; j < 2; ++j) {
                    v16bf bk = load_frag_b(Ks + (fn0 + j) * 16 * 64 + d0, 64);
                    sacc[j] = __builtin_amdgcn_wmma_f32_16x16x32_bf16(
                        false, aq, false, bk, (short)0, sacc[j], false, false);
                }
            }
#pragma unroll
            for (int j = 0; j < 2; ++j)
#pragma unroll
                for (int r = 0; r < 8; ++r)
                    Ss[(fm * 16 + r + 8 * hf) * 64 + (fn0 + j) * 16 + nl] = sacc[j][r];
        }
        __syncthreads();

        // online softmax update: 4 threads per row, 16 cols each
        {
            int row = tid >> 2, q4 = tid & 3;
            int tg = t0 + row;
            const int4* mrow = (const int4*)(mask + ((size_t)b * T + tg) * T + s0 + q4 * 16);
            float vals[16];
            float mx = -3.0e38f;
#pragma unroll
            for (int i4 = 0; i4 < 4; ++i4) {
                int4 mm = mrow[i4];
                int mks[4] = { mm.x, mm.y, mm.z, mm.w };
#pragma unroll
                for (int u = 0; u < 4; ++u) {
                    int i = i4 * 4 + u;
                    float v = Ss[row * 64 + q4 * 16 + i];
                    v = mks[u] ? v * scale : -1.0e9f;
                    vals[i] = v;
                    mx = fmaxf(mx, v);
                }
            }
            red[row * 4 + q4] = mx;
            __syncthreads();
            if (q4 == 0) {
                float tm = fmaxf(fmaxf(red[row * 4], red[row * 4 + 1]),
                                 fmaxf(red[row * 4 + 2], red[row * 4 + 3]));
                float oldm = rowm[row];
                float newm = fmaxf(oldm, tm);
                float alpha = __expf(oldm - newm);
                rowalpha[row] = alpha;
                rowm[row] = newm;
                rowl[row] *= alpha;
            }
            __syncthreads();
            float newm = rowm[row];
            float lsum = 0.0f;
#pragma unroll
            for (int i = 0; i < 16; ++i) {
                float p = __expf(vals[i] - newm);
                lsum += p;
                Ps[row * 64 + q4 * 16 + i] = f2bf(p);
            }
            red[row * 4 + q4] = lsum;
            __syncthreads();
            if (q4 == 0)
                rowl[row] += red[row * 4] + red[row * 4 + 1] +
                             red[row * 4 + 2] + red[row * 4 + 3];
        }
        __syncthreads();

        // O = O*alpha + P @ V
#pragma unroll
        for (int j = 0; j < 2; ++j)
#pragma unroll
            for (int r = 0; r < 8; ++r)
                o[j][r] *= rowalpha[fm * 16 + r + 8 * hf];
#pragma unroll
        for (int k0 = 0; k0 < 64; k0 += 32) {
            v16bf ap = load_frag_a(Ps + fm * 16 * 64 + k0, 64);
#pragma unroll
            for (int j = 0; j < 2; ++j) {
                v16bf bv = load_frag_b(VsT + (fn0 + j) * 16 * 64 + k0, 64);
                o[j] = __builtin_amdgcn_wmma_f32_16x16x32_bf16(
                    false, ap, false, bv, (short)0, o[j], false, false);
            }
        }
    }
    __syncthreads();

    // normalize and store to X[b][t][c][h*64+d] (bf16)
#pragma unroll
    for (int j = 0; j < 2; ++j)
#pragma unroll
        for (int r = 0; r < 8; ++r) {
            int row = fm * 16 + r + 8 * hf;
            int t = t0 + row;
            int d = (fn0 + j) * 16 + nl;
            float v = o[j][r] / rowl[row];
            X[(((size_t)b * T + t) * 8 + c) * 512 + h * 64 + d] = f2bf(v);
        }
}

// ---------------- host launch ----------------
extern "C" void kernel_launch(void* const* d_in, const int* in_sizes, int n_in,
                              void* d_out, int out_size, void* d_ws, size_t ws_size,
                              hipStream_t stream) {
    constexpr size_t ACT = 2ull * 512 * 8 * 512;  // 4,194,304 elements
    constexpr size_t WSZ = 512ull * 512;          // 262,144 elements

    const float* q  = (const float*)d_in[0];
    const float* k  = (const float*)d_in[1];
    const float* v  = (const float*)d_in[2];
    const int*   mk = (const int*)  d_in[3];
    const float* Wq = (const float*)d_in[4];
    const float* bq = (const float*)d_in[5];
    const float* Wk = (const float*)d_in[6];
    const float* bk = (const float*)d_in[7];
    const float* Wv = (const float*)d_in[8];
    const float* bv = (const float*)d_in[9];
    const float* Wo = (const float*)d_in[10];
    const float* bo = (const float*)d_in[11];

    bf16* qin = (bf16*)d_ws;
    bf16* kin = qin + ACT;
    bf16* vin = kin + ACT;
    bf16* wqt = vin + ACT;     // transposed bf16 weights: WT[n][k]
    bf16* wkt = wqt + WSZ;
    bf16* wvt = wkt + WSZ;
    bf16* wot = wvt + WSZ;
    bf16* qp  = wot + WSZ;
    bf16* kp  = qp + ACT;
    bf16* vp  = kp + ACT;      // V projection stored [b][h][c][dk][t]
    bf16* xb  = vp + ACT;

    const int CT = 256;
    cvt_f32_bf16<<<(int)(ACT / 4 / CT), CT, 0, stream>>>(q, qin, (int)ACT);
    cvt_f32_bf16<<<(int)(ACT / 4 / CT), CT, 0, stream>>>(k, kin, (int)ACT);
    cvt_f32_bf16<<<(int)(ACT / 4 / CT), CT, 0, stream>>>(v, vin, (int)ACT);

    dim3 tg(16, 16);
    cvt_transpose_f32_bf16<<<tg, 256, 0, stream>>>(Wq, wqt);
    cvt_transpose_f32_bf16<<<tg, 256, 0, stream>>>(Wk, wkt);
    cvt_transpose_f32_bf16<<<tg, 256, 0, stream>>>(Wv, wvt);
    cvt_transpose_f32_bf16<<<tg, 256, 0, stream>>>(Wo, wot);

    dim3 gg(8192 / 128, 512 / 128);  // 64 x 4 blocks
    gemm_bf16<<<gg, 256, 0, stream>>>(qin, wqt, bq, qp, 0);
    gemm_bf16<<<gg, 256, 0, stream>>>(kin, wkt, bk, kp, 0);
    gemm_bf16<<<gg, 256, 0, stream>>>(vin, wvt, bv, vp, 2);

    attn_kernel<<<1024, 256, 0, stream>>>(qp, kp, vp, mk, xb);

    gemm_bf16<<<gg, 256, 0, stream>>>(xb, wot, bo, d_out, 1);
}
